// DTree_66254165508708
// MI455X (gfx1250) — matrix operand order repaired and checked by reading, not compiled, not run on hardware
//
#include <hip/hip_runtime.h>
#include <math.h>

// ---------- types ----------
typedef __attribute__((ext_vector_type(16))) __bf16   v16bf;
typedef __attribute__((ext_vector_type(8)))  float    v8f;
typedef __attribute__((ext_vector_type(4)))  unsigned int u32x4;

// int4-vector type matching the async-to-LDS builtin's parameter type
typedef int v4i __attribute__((vector_size(16)));
typedef __attribute__((address_space(1))) v4i GV4;   // global
typedef __attribute__((address_space(3))) v4i LV4;   // LDS

union Frag { u32x4 q[2]; v16bf v; };

static __device__ inline v8f wmma_bf16(v16bf a, v16bf b, v8f c) {
    return __builtin_amdgcn_wmma_f32_16x16x32_bf16(false, a, false, b, (short)0, c, false, false);
}

// replicate bf16(f) into both halves of a dword for v_pk_mul_bf16
static __device__ inline unsigned bfpair(float f) {
    __bf16 b = (__bf16)f;
    unsigned short u = __builtin_bit_cast(unsigned short, b);
    return (unsigned)u * 0x10001u;
}

// frag *= splat(scale) using packed bf16 multiplies (8 VALU ops)
static __device__ inline v16bf pkscale(Frag f, unsigned s2) {
    Frag r;
    unsigned* o = (unsigned*)&r;
    const unsigned* a = (const unsigned*)&f;
#pragma unroll
    for (int i = 0; i < 8; ++i)
        asm("v_pk_mul_bf16 %0, %1, %2" : "=v"(o[i]) : "v"(a[i]), "v"(s2));
    return r.v;
}

// ---------- problem sizes ----------
#define BATCH   8192
#define D_IN    512
#define D1      513      // D_IN + 1
#define KP      544      // D1 padded to 17*32 (col 513 = bias lane)
#define NNODES  127
#define NLEAF   128
#define D_OUT   256
#define NROUTE  7        // DEPTH+1

// ---------- workspace layout (bytes, 256-aligned) ----------
#define OFF_H      0UL                               // bf16 8192*544
#define OFF_WB     8912896UL                         // bf16 32768*544
#define OFF_XB     44564480UL                        // bf16 8192*512
#define OFF_WPREB  52953088UL                        // bf16 513*512
#define OFF_NW     53478400UL                        // f32 127*513
#define OFF_SW     53739264UL                        // f32 8192*128
#define OFF_INVN   57933568UL                        // f32 8192
#define OFF_ENT    57966336UL                        // f32 scalar
#define OFF_SCALE  57966592UL                        // f32 scalar

// ===================== conversion kernels =====================
__global__ void k_cvt_x(const float* __restrict__ x, __bf16* __restrict__ xb,
                        float* __restrict__ ent, int n) {
    int i = blockIdx.x * 256 + threadIdx.x;
    if (i == 0) *ent = 0.0f;
    if (i < n) xb[i] = (__bf16)x[i];
}

__global__ void k_cvt_wpre(const float* __restrict__ w, __bf16* __restrict__ wb, int n) {
    int i = blockIdx.x * 256 + threadIdx.x;
    if (i < n) wb[i] = (__bf16)w[i];
}

__global__ void k_cvt_wleaf(const float* __restrict__ W, const float* __restrict__ bl,
                            __bf16* __restrict__ Wb) {
    for (int row = blockIdx.x; row < D_OUT * NLEAF; row += gridDim.x) {
        for (int k = threadIdx.x; k < KP; k += 256) {
            float v = (k < D1) ? W[(size_t)row * D1 + k]
                               : ((k == D1) ? bl[row] : 0.0f);
            Wb[(size_t)row * KP + k] = (__bf16)v;
        }
    }
}

__global__ void k_norm_w(const float* __restrict__ rw, float* __restrict__ nw) {
    __shared__ float red[256];
    int row = blockIdx.x, t = threadIdx.x;
    float s = 0.0f;
    for (int k = t; k < D1; k += 256) { float v = rw[(size_t)row * D1 + k]; s += v * v; }
    red[t] = s; __syncthreads();
    for (int st = 128; st > 0; st >>= 1) { if (t < st) red[t] += red[t + st]; __syncthreads(); }
    float inv = 1.0f / fmaxf(sqrtf(red[0]), 1e-12f);
    for (int k = t; k < D1; k += 256) nw[(size_t)row * D1 + k] = rw[(size_t)row * D1 + k] * inv;
}

// ===================== pre-GEMM: h = relu(x @ W_pre^T + b_pre) =====================
__global__ void k_pre_gemm(const __bf16* __restrict__ xb, const __bf16* __restrict__ wp,
                           const float* __restrict__ bpre, __bf16* __restrict__ h) {
    const int t = threadIdx.x, wave = t >> 5, lane = t & 31;
    const int lh = lane >> 4, ln = lane & 15;
    const int mrow = blockIdx.x * 64 + wave * 16;
    const int nb = blockIdx.y * 64;

    v8f acc[4];
    const v8f z = {0.f,0.f,0.f,0.f,0.f,0.f,0.f,0.f};
#pragma unroll
    for (int nt = 0; nt < 4; ++nt) acc[nt] = z;

    for (int kc = 0; kc < 16; ++kc) {
        const __bf16* ab = xb + (size_t)(mrow + ln) * D_IN + kc * 32 + lh * 8;
        Frag fa;
        fa.q[0] = *(const u32x4*)(ab);
        fa.q[1] = *(const u32x4*)(ab + 16);
#pragma unroll
        for (int nt = 0; nt < 4; ++nt) {
            int o = nb + nt * 16 + ln;
            int oc = o > 512 ? 512 : o;                 // clamp (tail tile)
            const __bf16* bb = wp + (size_t)oc * D_IN + kc * 32 + lh * 8;
            Frag fb;
            fb.q[0] = *(const u32x4*)(bb);
            fb.q[1] = *(const u32x4*)(bb + 16);
            acc[nt] = wmma_bf16(fa.v, fb.v, acc[nt]);
        }
    }
#pragma unroll
    for (int nt = 0; nt < 4; ++nt) {
#pragma unroll
        for (int r = 0; r < 8; ++r) {
            int col = nb + nt * 16 + ln;
            int row = mrow + r + lh * 8;
            size_t idx = (size_t)row * KP + col;
            if (col < D1) {
                float v = acc[nt][r] + bpre[col];
                h[idx] = (__bf16)fmaxf(v, 0.0f);
            } else if (col == D1) {
                h[idx] = (__bf16)1.0f;
            } else if (col < KP) {
                h[idx] = (__bf16)0.0f;
            }
        }
    }
}

// ===================== row norms of h =====================
__global__ void k_rownorm(const __bf16* __restrict__ h, float* __restrict__ invn) {
    __shared__ float red[128];
    int b = blockIdx.x, t = threadIdx.x;
    float s = 0.0f;
    for (int k = t; k < D1; k += 128) { float v = (float)h[(size_t)b * KP + k]; s += v * v; }
    red[t] = s; __syncthreads();
    for (int st = 64; st > 0; st >>= 1) { if (t < st) red[t] += red[t + st]; __syncthreads(); }
    if (t == 0) invn[b] = 1.0f / fmaxf(sqrtf(red[0]), 1e-12f);
}

// ===================== routing: s_w + entropy =====================
__global__ void k_route(const __bf16* __restrict__ h, const float* __restrict__ invn,
                        const float* __restrict__ nw, const int* __restrict__ ridx,
                        const int* __restrict__ rside, float* __restrict__ sw,
                        float* __restrict__ ent_sum) {
    __shared__ float hrow[D1];
    __shared__ float rdist[NLEAF];
    __shared__ float red[128];
    const int b = blockIdx.x, t = threadIdx.x;
    for (int k = t; k < D1; k += 128) hrow[k] = (float)h[(size_t)b * KP + k];
    __syncthreads();
    const float inv = invn[b];
    if (t < NNODES) {
        float acc = 0.0f;
        for (int k = 0; k < D1; ++k) acc += hrow[k] * nw[(size_t)t * D1 + k];
        float right = acc * inv;
        rdist[t] = (1.0f - right) * 0.5f;
    }
    __syncthreads();
    float lp = 0.0f;
#pragma unroll
    for (int j = 0; j < NROUTE; ++j) {
        int node = ridx[t * NROUTE + j];
        int side = rside[t * NROUTE + j];
        float p = (side == 0) ? rdist[node] : 1.0f - rdist[node];
        p = fminf(fmaxf(p, 0.01f), 0.99f);
        lp += logf(p);
    }
    float s = expf(lp);
    sw[(size_t)b * NLEAF + t] = s;
    red[t] = -s * lp;
    __syncthreads();
    for (int st = 64; st > 0; st >>= 1) { if (t < st) red[t] += red[t + st]; __syncthreads(); }
    if (t == 0) atomicAdd(ent_sum, red[0]);
}

__global__ void k_scale(const float* __restrict__ ent, float* __restrict__ sc) {
    const float maxent = (128.0f / 6.0f) * logf(6.0f);
    *sc = 1.0f + (*ent / (float)BATCH) / maxent;
}

// ===================== fused leaf GEMM =====================
// out[b,:] = scale * sum_l (s_w[l,b] * h[b,:]) @ W_leaf[l]^T   (bias via K=513 lane)
// 128 threads (4 wave32). Tile M=32 (2 WMMA M-tiles) x N=256 (64/wave). K=544 (17 chunks).
// B stream double-buffered across K-chunks/leaves so loads overlap WMMA.
__global__ void __launch_bounds__(128)
k_leaf_gemm(const __bf16* __restrict__ h, const __bf16* __restrict__ Wb,
            const float* __restrict__ sw, const float* __restrict__ scale_p,
            float* __restrict__ out) {
    __shared__ __bf16 hs[32 * KP];       // 34816 B
    __shared__ float  sws[32 * NLEAF];   // 16384 B
    const int t = threadIdx.x, wave = t >> 5, lane = t & 31;
    const int lh = lane >> 4, ln = lane & 15;
    const int Mbase = blockIdx.x * 32;
    const int nbase = wave * 64;

    { // stage A tile and s_w tile into LDS (CDNA5 async-to-LDS path)
#if __has_builtin(__builtin_amdgcn_global_load_async_to_lds_b128)
        GV4* gsrc = (GV4*)(h + (size_t)Mbase * KP);
        LV4* ldst = (LV4*)hs;
        for (int i = t; i < 32 * KP * 2 / 16; i += 128)
            __builtin_amdgcn_global_load_async_to_lds_b128(gsrc + i, ldst + i, 0, 0);
        GV4* gsw = (GV4*)(sw + (size_t)Mbase * NLEAF);
        LV4* lsw = (LV4*)sws;
        for (int i = t; i < 32 * NLEAF * 4 / 16; i += 128)
            __builtin_amdgcn_global_load_async_to_lds_b128(gsw + i, lsw + i, 0, 0);
#if __has_builtin(__builtin_amdgcn_s_wait_asynccnt)
        __builtin_amdgcn_s_wait_asynccnt(0);
#else
        asm volatile("s_wait_asynccnt 0x0" ::: "memory");
#endif
#else
        const u32x4* src = (const u32x4*)(h + (size_t)Mbase * KP);
        u32x4* dst = (u32x4*)hs;
        for (int i = t; i < 32 * KP / 8; i += 128) dst[i] = src[i];
        const u32x4* s2 = (const u32x4*)(sw + (size_t)Mbase * NLEAF);
        u32x4* d2 = (u32x4*)sws;
        for (int i = t; i < 32 * NLEAF / 4; i += 128) d2[i] = s2[i];
#endif
    }
    __syncthreads();

    v8f acc[2][4];
    const v8f z = {0.f,0.f,0.f,0.f,0.f,0.f,0.f,0.f};
#pragma unroll
    for (int mt = 0; mt < 2; ++mt)
#pragma unroll
        for (int nt = 0; nt < 4; ++nt) acc[mt][nt] = z;

    unsigned s2a = 0, s2b = 0;

    // load the 4 B fragments for (leaf ll, chunk kcc) into dst[0..3]
    auto ldb = [&](int ll, int kcc, Frag* dst) {
#pragma unroll
        for (int nt = 0; nt < 4; ++nt) {
            const __bf16* bb = Wb + ((size_t)ll * D_OUT + nbase + nt * 16 + ln) * KP
                               + kcc * 32 + lh * 8;
            dst[nt].q[0] = *(const u32x4*)(bb);
            dst[nt].q[1] = *(const u32x4*)(bb + 16);
        }
    };
    // one K-chunk of compute against buffered B frags
    auto step = [&](int kc, Frag* cur) {
        const int kb = kc * 32 + lh * 8;
        Frag fa0, fa1;
        fa0.q[0] = *(const u32x4*)(hs + (size_t)ln * KP + kb);
        fa0.q[1] = *(const u32x4*)(hs + (size_t)ln * KP + kb + 16);
        fa1.q[0] = *(const u32x4*)(hs + (size_t)(16 + ln) * KP + kb);
        fa1.q[1] = *(const u32x4*)(hs + (size_t)(16 + ln) * KP + kb + 16);
        const v16bf a0 = pkscale(fa0, s2a);
        const v16bf a1 = pkscale(fa1, s2b);
#pragma unroll
        for (int nt = 0; nt < 4; ++nt) {
            acc[0][nt] = wmma_bf16(a0, cur[nt].v, acc[0][nt]);
            acc[1][nt] = wmma_bf16(a1, cur[nt].v, acc[1][nt]);
        }
    };

    Frag fb0[4], fb1[4];
    ldb(0, 0, fb0);                       // prime pipeline

    for (int l = 0; l < NLEAF; ++l) {
        if (l + 1 < NLEAF)                // warm L2 for next leaf's B rows
            __builtin_prefetch(Wb + ((size_t)(l + 1) * D_OUT + t) * KP, 0, 1);

        s2a = bfpair(sws[(0  + ln) * NLEAF + l]);
        s2b = bfpair(sws[(16 + ln) * NLEAF + l]);

#pragma unroll
        for (int p = 0; p < 8; ++p) {     // kc pairs: (0,1)..(14,15)
            ldb(l, 2 * p + 1, fb1);
            step(2 * p, fb0);
            ldb(l, 2 * p + 2, fb0);       // 2p+2 <= 16
            step(2 * p + 1, fb1);
        }
        // tail: fb0 holds (l,16); preload next leaf's first chunk
        int lnx = (l + 1 < NLEAF) ? l + 1 : NLEAF - 1;
        ldb(lnx, 0, fb1);
        step(16, fb0);
#pragma unroll
        for (int q = 0; q < 4; ++q) fb0[q] = fb1[q];
    }

    const float sc = scale_p[0];
#pragma unroll
    for (int mt = 0; mt < 2; ++mt)
#pragma unroll
        for (int nt = 0; nt < 4; ++nt)
#pragma unroll
            for (int r = 0; r < 8; ++r) {
                int row = Mbase + mt * 16 + r + lh * 8;
                int col = nbase + nt * 16 + ln;
                out[(size_t)row * D_OUT + col] = acc[mt][nt][r] * sc;
            }
}

// ===================== launch =====================
extern "C" void kernel_launch(void* const* d_in, const int* in_sizes, int n_in,
                              void* d_out, int out_size, void* d_ws, size_t ws_size,
                              hipStream_t stream) {
    const float* x       = (const float*)d_in[0];
    const float* W_pre   = (const float*)d_in[1];
    const float* b_pre   = (const float*)d_in[2];
    const float* right_w = (const float*)d_in[3];
    const float* W_leaf  = (const float*)d_in[4];
    const float* b_leaf  = (const float*)d_in[5];
    const int*   ridx    = (const int*)d_in[6];
    const int*   rside   = (const int*)d_in[7];
    float* out = (float*)d_out;

    char* ws = (char*)d_ws;
    __bf16* h_bf   = (__bf16*)(ws + OFF_H);
    __bf16* Wb     = (__bf16*)(ws + OFF_WB);
    __bf16* xb     = (__bf16*)(ws + OFF_XB);
    __bf16* wpreb  = (__bf16*)(ws + OFF_WPREB);
    float*  nw     = (float*)(ws + OFF_NW);
    float*  sw     = (float*)(ws + OFF_SW);
    float*  invn   = (float*)(ws + OFF_INVN);
    float*  ent    = (float*)(ws + OFF_ENT);
    float*  scale  = (float*)(ws + OFF_SCALE);

    k_cvt_x<<<(BATCH * D_IN + 255) / 256, 256, 0, stream>>>(x, xb, ent, BATCH * D_IN);
    k_cvt_wpre<<<(D1 * D_IN + 255) / 256, 256, 0, stream>>>(W_pre, wpreb, D1 * D_IN);
    k_cvt_wleaf<<<2048, 256, 0, stream>>>(W_leaf, b_leaf, Wb);
    k_norm_w<<<NNODES, 256, 0, stream>>>(right_w, nw);

    k_pre_gemm<<<dim3(BATCH / 64, 9), 128, 0, stream>>>(xb, wpreb, b_pre, h_bf);
    k_rownorm<<<BATCH, 128, 0, stream>>>(h_bf, invn);
    k_route<<<BATCH, 128, 0, stream>>>(h_bf, invn, nw, ridx, rside, sw, ent);
    k_scale<<<1, 1, 0, stream>>>(ent, scale);

    k_leaf_gemm<<<BATCH / 32, 128, 0, stream>>>(h_bf, Wb, sw, scale, out);
}
